// MQA_23390391894531
// MI455X (gfx1250) — matrix operand order repaired and checked by reading, not compiled, not run on hardware
//
#include <hip/hip_runtime.h>

// ---------------- constants from the reference ----------------
#define B_ 2
#define T_ 2048
#define C_ 2048
#define H_ 16
#define D_ 128
#define R_ (B_ * T_)   // 4096 total token rows

typedef __bf16 bf16_t;
typedef __attribute__((ext_vector_type(16))) __bf16 v16bf;
typedef __attribute__((ext_vector_type(8)))  __bf16 v8bf;
typedef __attribute__((ext_vector_type(8)))  float  v8f;
typedef __attribute__((ext_vector_type(4)))  unsigned int u32x4;
typedef __attribute__((ext_vector_type(8)))  int          i32x8;
typedef __attribute__((ext_vector_type(4)))  int          i32x4;

// ---------------- helpers ----------------
static __device__ __forceinline__ bf16_t f2bf(float f) {
  unsigned u = __builtin_bit_cast(unsigned, f);
  unsigned short h;
  if ((u & 0x7F800000u) == 0x7F800000u) {                // Inf / NaN
    h = (unsigned short)((u >> 16) | ((u & 0xFFFFu) ? 0x40u : 0u));
  } else {                                               // round-to-nearest-even
    unsigned r = u + 0x7FFFu + ((u >> 16) & 1u);
    h = (unsigned short)(r >> 16);
  }
  return __builtin_bit_cast(bf16_t, h);
}

// Build a v16bf A/B fragment from two contiguous 16-byte chunks.
static __device__ __forceinline__ v16bf load2x8(const bf16_t* p0, const bf16_t* p1) {
  v8bf lo = *reinterpret_cast<const v8bf*>(p0);
  v8bf hi = *reinterpret_cast<const v8bf*>(p1);
  return __builtin_shufflevector(lo, hi, 0,1,2,3,4,5,6,7,8,9,10,11,12,13,14,15);
}

static __device__ __forceinline__ v8f wmma_bf16(v16bf a, v16bf b, v8f c) {
  return __builtin_amdgcn_wmma_f32_16x16x32_bf16(false, a, false, b, (short)0, c,
                                                 false, false);
}

// ---------------- Tensor Data Mover: 2D bf16 tile -> LDS ----------------
// D# packing per CDNA5 ISA 8.3/8.4: count=1, type=2 ("image"), data_size=1 (2B),
// tile = tile_d0 x tile_d1, row stride = stride0 (elements), no padding/iterate.
static __device__ __forceinline__ void tdm_load_tile_2d(
    unsigned lds_off, const void* gptr,
    unsigned tensor_d0, unsigned tensor_d1,
    unsigned tile_d0, unsigned tile_d1,
    unsigned long long stride0) {
  unsigned long long ga = (unsigned long long)(size_t)gptr;
  u32x4 g0;
  g0[0] = 1u;                                             // count=1, user desc
  g0[1] = lds_off;                                        // lds_addr (bytes)
  g0[2] = (unsigned)(ga & 0xFFFFFFFFu);                   // global_addr[31:0]
  g0[3] = (unsigned)((ga >> 32) & 0x01FFFFFFu) | 0x80000000u;  // ga[56:32]|type=2
  i32x8 g1;
  g1[0] = (int)(1u << 16);                                // data_size=1 (2 bytes)
  g1[1] = (int)((tensor_d0 & 0xFFFFu) << 16);             // abar=0 | td0[15:0]
  g1[2] = (int)(((tensor_d0 >> 16) & 0xFFFFu) | ((tensor_d1 & 0xFFFFu) << 16));
  g1[3] = (int)(((tensor_d1 >> 16) & 0xFFFFu) | ((tile_d0 & 0xFFFFu) << 16));
  g1[4] = (int)(tile_d1 & 0xFFFFu);                       // tile_d1 | tile_d2=0
  g1[5] = (int)(unsigned)(stride0 & 0xFFFFFFFFu);         // dim0_stride[31:0]
  g1[6] = (int)(unsigned)((stride0 >> 32) & 0xFFFFu);     // dim0_stride[47:32]
  g1[7] = 0;
  i32x4 z4; z4[0] = 0; z4[1] = 0; z4[2] = 0; z4[3] = 0;   // 2D: groups 2/3 unused
#if __has_include(<hip/amd_detail/amd_gfx1250_TDM.h>)
  i32x8 z8; z8[0]=0; z8[1]=0; z8[2]=0; z8[3]=0; z8[4]=0; z8[5]=0; z8[6]=0; z8[7]=0;
  __builtin_amdgcn_tensor_load_to_lds(g0, g1, z4, z4, z8, 0);
#else
  __builtin_amdgcn_tensor_load_to_lds(g0, g1, z4, z4, 0);
#endif
}

// ---------------- elementwise convert fp32 -> bf16 ----------------
__global__ __launch_bounds__(256)
void cvt_bf16_kernel(const float* __restrict__ in, bf16_t* __restrict__ out, int n) {
  int i = blockIdx.x * 256 + threadIdx.x;
  if (i < n) out[i] = f2bf(in[i]);
}

// ---------------- transpose fp32 [K][N] -> bf16 [N][K] ----------------
__global__ __launch_bounds__(256)
void transpose_bf16_kernel(const float* __restrict__ W, bf16_t* __restrict__ Wt,
                           int K, int N) {
  int i = blockIdx.x * 256 + threadIdx.x;
  if (i < K * N) {
    int k = i / N, n = i - k * N;
    Wt[(size_t)n * K + k] = f2bf(W[i]);
  }
}

// ---------------- TDM-fed LDS-tiled bf16 WMMA GEMM -----------------------
// C[MxN] = A[MxK] * Bt[NxK]^T.  Block = 8 waves = 128x128 C tile; K-step 32.
// Wave 0 drives the Tensor Data Mover: A panel 128x32 and Bt panel 128x32
// (8 KB each) double-buffered in 32 KB dynamic LDS; everyone computes from LDS.
// mode 0: fp32 row-major ; mode 1: bf16 row-major ; mode 2: bf16 transposed.
#define STAGE_BYTES 16384            // 8KB A + 8KB B per stage
__global__ __launch_bounds__(256)
void gemm_bf16_wmma_kernel(const bf16_t* __restrict__ A, const bf16_t* __restrict__ Bt,
                           float* __restrict__ Cf, bf16_t* __restrict__ Cb,
                           int M, int N, int K, int mode, int ldct) {
  extern __shared__ __align__(16) char smem[];
  const int lane = threadIdx.x & 31;
  const int wave = threadIdx.x >> 5;
  const int m0 = blockIdx.x << 7;            // 128-row block tile
  const int n0 = blockIdx.y << 7;            // 128-col block tile
  const int wm = wave >> 1;                  // 0..3 : 32-row strip
  const int wn = wave & 1;                   // 0..1 : 64-col strip
  const int lr = lane & 15, hi = lane >> 4;

  v8f acc[2][4];
#pragma unroll
  for (int u = 0; u < 2; ++u)
#pragma unroll
    for (int t = 0; t < 4; ++t)
#pragma unroll
      for (int j = 0; j < 8; ++j) acc[u][t][j] = 0.0f;

  const int nk = K >> 5;                     // K-steps of 32
  if (wave == 0) {
    tdm_load_tile_2d(0u,        A  + (size_t)m0 * K, K, M, 32u, 128u, (unsigned long long)K);
    tdm_load_tile_2d(8192u,     Bt + (size_t)n0 * K, K, N, 32u, 128u, (unsigned long long)K);
  }

  for (int ki = 0; ki < nk; ++ki) {
    const int cur = ki & 1;
    if (wave == 0) {
      if (ki + 1 < nk) {
        const int nxt = cur ^ 1;
        const int kk = (ki + 1) << 5;
        tdm_load_tile_2d((unsigned)(nxt * STAGE_BYTES),
                         A + (size_t)m0 * K + kk, K, M, 32u, 128u,
                         (unsigned long long)K);
        tdm_load_tile_2d((unsigned)(nxt * STAGE_BYTES + 8192),
                         Bt + (size_t)n0 * K + kk, K, N, 32u, 128u,
                         (unsigned long long)K);
        __builtin_amdgcn_s_wait_tensorcnt(2);   // stage `cur` fully landed
      } else {
        __builtin_amdgcn_s_wait_tensorcnt(0);
      }
    }
    __syncthreads();                            // staged panels visible to block

    const bf16_t* As = reinterpret_cast<const bf16_t*>(smem + cur * STAGE_BYTES);
    const bf16_t* Bs = reinterpret_cast<const bf16_t*>(smem + cur * STAGE_BYTES + 8192);

    v16bf afr[2], bfr[4];
#pragma unroll
    for (int u = 0; u < 2; ++u) {
      const bf16_t* ar = As + (size_t)(wm * 32 + u * 16 + lr) * 32;
      afr[u] = load2x8(ar + 16 * hi, ar + 16 * hi + 8);
    }
#pragma unroll
    for (int t = 0; t < 4; ++t) {
      const bf16_t* br = Bs + (size_t)(wn * 64 + t * 16 + lr) * 32 + 16 * hi;
      bfr[t] = load2x8(br, br + 8);
    }
#pragma unroll
    for (int u = 0; u < 2; ++u)
#pragma unroll
      for (int t = 0; t < 4; ++t)
        acc[u][t] = wmma_bf16(afr[u], bfr[t], acc[u][t]);

    __syncthreads();                            // done reading before re-fill
  }

#pragma unroll
  for (int u = 0; u < 2; ++u)
#pragma unroll
    for (int t = 0; t < 4; ++t) {
      const int n = n0 + wn * 64 + t * 16 + lr;
#pragma unroll
      for (int j = 0; j < 8; ++j) {
        const int m = m0 + wm * 32 + u * 16 + j + 8 * hi;
        const float v = acc[u][t][j];
        if (mode == 0)      Cf[(size_t)m * N + n] = v;
        else if (mode == 1) Cb[(size_t)m * N + n] = f2bf(v);
        else                Cb[(size_t)n * ldct + m] = f2bf(v);
      }
    }
}

// ---------------- flash MQA attention ----------------
// Q  : [R][C] bf16 (per head h use cols h*128..h*128+127)
// Kb : [R][D] bf16 (row-major == Bt form for Q*K^T)
// Vt : [D][R] bf16 (d-major == Bt form for P*V)
// Y  : [R][C] bf16  (y concatenated back to [b,t,h*d])
__global__ __launch_bounds__(128)
void mqa_attention_kernel(const bf16_t* __restrict__ Q, const bf16_t* __restrict__ Kb,
                          const bf16_t* __restrict__ Vt, bf16_t* __restrict__ Y) {
  __shared__ bf16_t plds[4][16 * 32];          // per-wave P staging (C->A transpose)
  const int lane = threadIdx.x & 31;
  const int wave = threadIdx.x >> 5;
  const int wid  = blockIdx.x * 4 + wave;      // 0 .. B*H*(T/16)-1
  const int qt = wid % (T_ / 16);
  const int h  = (wid / (T_ / 16)) % H_;
  const int b  = wid / ((T_ / 16) * H_);
  const int lr = lane & 15, hi = lane >> 4;

  // Q fragments (16 queries x 128 d) as 4 A-tiles of 16x32
  const bf16_t* qrow = Q + (size_t)(b * T_ + qt * 16 + lr) * C_ + h * D_;
  v16bf qa[4];
#pragma unroll
  for (int kt = 0; kt < 4; ++kt)
    qa[kt] = load2x8(qrow + kt * 32 + 16 * hi, qrow + kt * 32 + 16 * hi + 8);

  float m[8], l[8];
  v8f o[8];
#pragma unroll
  for (int j = 0; j < 8; ++j) { m[j] = -1e30f; l[j] = 0.0f; }
#pragma unroll
  for (int dt = 0; dt < 8; ++dt)
#pragma unroll
    for (int j = 0; j < 8; ++j) o[dt][j] = 0.0f;

  const float scale = 0.08838834764831845f;    // 1/sqrt(128)
  bf16_t* P = plds[wave];

  for (int s0 = 0; s0 < T_; s0 += 32) {
    // ---- scores: two 16x16 tiles (keys s0..s0+15 and s0+16..s0+31) ----
    v8f sc0, sc1;
#pragma unroll
    for (int j = 0; j < 8; ++j) { sc0[j] = 0.0f; sc1[j] = 0.0f; }
    const bf16_t* krow0 = Kb + (size_t)(b * T_ + s0 + lr) * D_;
    const bf16_t* krow1 = krow0 + (size_t)16 * D_;
#pragma unroll
    for (int kt = 0; kt < 4; ++kt) {
      v16bf b0 = load2x8(krow0 + kt * 32 + 16 * hi, krow0 + kt * 32 + 16 * hi + 8);
      sc0 = wmma_bf16(qa[kt], b0, sc0);
      v16bf b1 = load2x8(krow1 + kt * 32 + 16 * hi, krow1 + kt * 32 + 16 * hi + 8);
      sc1 = wmma_bf16(qa[kt], b1, sc1);
    }

    // ---- online softmax (rows striped: row j+8*hi, cols across 16-lane half) ----
    float p0[8], p1[8], sf[8];
#pragma unroll
    for (int j = 0; j < 8; ++j) {
      float x0 = sc0[j] * scale, x1 = sc1[j] * scale;
      float mx = fmaxf(x0, x1);
      mx = fmaxf(mx, __shfl_xor(mx, 1, 32));
      mx = fmaxf(mx, __shfl_xor(mx, 2, 32));
      mx = fmaxf(mx, __shfl_xor(mx, 4, 32));
      mx = fmaxf(mx, __shfl_xor(mx, 8, 32));
      float mnew = fmaxf(m[j], mx);
      sf[j] = __expf(m[j] - mnew);
      float e0 = __expf(x0 - mnew), e1 = __expf(x1 - mnew);
      float rs = e0 + e1;
      rs += __shfl_xor(rs, 1, 32);
      rs += __shfl_xor(rs, 2, 32);
      rs += __shfl_xor(rs, 4, 32);
      rs += __shfl_xor(rs, 8, 32);
      l[j] = l[j] * sf[j] + rs;
      m[j] = mnew;
      p0[j] = e0; p1[j] = e1;
    }
#pragma unroll
    for (int dt = 0; dt < 8; ++dt)
#pragma unroll
      for (int j = 0; j < 8; ++j) o[dt][j] *= sf[j];

    // ---- P: C-layout -> A-layout via LDS (16 rows x 32 cols bf16) ----
#pragma unroll
    for (int j = 0; j < 8; ++j) {
      P[(j + 8 * hi) * 32 + lr]      = f2bf(p0[j]);
      P[(j + 8 * hi) * 32 + 16 + lr] = f2bf(p1[j]);
    }
    __builtin_amdgcn_fence(__ATOMIC_RELEASE, "workgroup");
    __builtin_amdgcn_wave_barrier();
    __builtin_amdgcn_fence(__ATOMIC_ACQUIRE, "workgroup");
    const bf16_t* prow = P + lr * 32;
    v16bf pa = load2x8(prow + 16 * hi, prow + 16 * hi + 8);
    __builtin_amdgcn_wave_barrier();

    // ---- o += P (16x32) * V (32x16 per d-tile), 8 d-tiles ----
#pragma unroll
    for (int dt = 0; dt < 8; ++dt) {
      const bf16_t* vrow = Vt + (size_t)(dt * 16 + lr) * R_ + b * T_ + s0 + 16 * hi;
      v16bf vb = load2x8(vrow, vrow + 8);
      o[dt] = wmma_bf16(pa, vb, o[dt]);
    }
  }

  // ---- normalize + store y back into [b,t, h*128 + d] bf16 ----
#pragma unroll
  for (int j = 0; j < 8; ++j) {
    const float inv = 1.0f / l[j];
    bf16_t* yrow = Y + (size_t)(b * T_ + qt * 16 + j + 8 * hi) * C_ + h * D_;
#pragma unroll
    for (int dt = 0; dt < 8; ++dt)
      yrow[dt * 16 + lr] = f2bf(o[dt][j] * inv);
  }
}

// ---------------- host-side orchestration ----------------
extern "C" void kernel_launch(void* const* d_in, const int* in_sizes, int n_in,
                              void* d_out, int out_size, void* d_ws, size_t ws_size,
                              hipStream_t stream) {
  const float* x  = (const float*)d_in[0];
  const float* Wq = (const float*)d_in[1];
  const float* Wk = (const float*)d_in[2];
  const float* Wv = (const float*)d_in[3];
  const float* Wo = (const float*)d_in[4];
  float* out = (float*)d_out;

  char* w = (char*)d_ws;
  bf16_t* xbf = (bf16_t*)w; w += (size_t)R_ * C_ * 2;
  bf16_t* WqT = (bf16_t*)w; w += (size_t)C_ * C_ * 2;
  bf16_t* WkT = (bf16_t*)w; w += (size_t)D_ * C_ * 2;
  bf16_t* WvT = (bf16_t*)w; w += (size_t)D_ * C_ * 2;
  bf16_t* WoT = (bf16_t*)w; w += (size_t)C_ * C_ * 2;
  bf16_t* Qbf = (bf16_t*)w; w += (size_t)R_ * C_ * 2;
  bf16_t* Kbf = (bf16_t*)w; w += (size_t)R_ * D_ * 2;
  bf16_t* Vtb = (bf16_t*)w; w += (size_t)D_ * R_ * 2;
  bf16_t* Ybf = (bf16_t*)w; w += (size_t)R_ * C_ * 2;

  { int n = R_ * C_; cvt_bf16_kernel<<<(n + 255) / 256, 256, 0, stream>>>(x, xbf, n); }
  { int n = C_ * C_; transpose_bf16_kernel<<<(n + 255) / 256, 256, 0, stream>>>(Wq, WqT, C_, C_); }
  { int n = C_ * D_; transpose_bf16_kernel<<<(n + 255) / 256, 256, 0, stream>>>(Wk, WkT, C_, D_); }
  { int n = C_ * D_; transpose_bf16_kernel<<<(n + 255) / 256, 256, 0, stream>>>(Wv, WvT, C_, D_); }
  { int n = C_ * C_; transpose_bf16_kernel<<<(n + 255) / 256, 256, 0, stream>>>(Wo, WoT, C_, C_); }

  auto gemm = [&](const bf16_t* A, const bf16_t* Bt, float* Cf, bf16_t* Cb,
                  int M, int N, int K, int mode, int ldct) {
    dim3 grid(M / 128, N / 128);
    gemm_bf16_wmma_kernel<<<grid, 256, 2 * STAGE_BYTES, stream>>>(A, Bt, Cf, Cb,
                                                                  M, N, K, mode, ldct);
  };

  gemm(xbf, WqT, nullptr, Qbf, R_, C_, C_, 1, 0);      // Q   = x Wq   (bf16)
  gemm(xbf, WkT, nullptr, Kbf, R_, D_, C_, 1, 0);      // K   = x Wk   (bf16 row-major)
  gemm(xbf, WvT, nullptr, Vtb, R_, D_, C_, 2, R_);     // V^T = (x Wv)^T  [D][R]

  mqa_attention_kernel<<<(B_ * H_ * (T_ / 16)) / 4, 128, 0, stream>>>(Qbf, Kbf, Vtb, Ybf);

  gemm(Ybf, WoT, out, nullptr, R_, C_, C_, 0, 0);      // out = y Wo  (fp32)
}
